// MemoryModule_40192303956202
// MI455X (gfx1250) — matrix operand order repaired and checked by reading, not compiled, not run on hardware
//
#include <hip/hip_runtime.h>
#include <hip/hip_bf16.h>
#include <math.h>

// ---------------------------------------------------------------------------
// Types for CDNA5 WMMA / TDM
// ---------------------------------------------------------------------------
typedef __attribute__((ext_vector_type(16))) __bf16 v16bf;
typedef __attribute__((ext_vector_type(8)))  float  v8f;
typedef __attribute__((ext_vector_type(4)))  unsigned int u32x4;
typedef __attribute__((ext_vector_type(8)))  int i32x8;
typedef __attribute__((ext_vector_type(4)))  int i32x4;

#define TILE_M 128
#define TILE_N 128
#define TILE_K 32
#define LDS_STRIDE 40   // ushorts per row (32 data + 8 pad) == TDM pad 16dw+4dw

__device__ __forceinline__ unsigned short f2bf(float f) {
    // round-to-nearest-even fp32 -> bf16
    unsigned u = __float_as_uint(f);
    u += 0x7FFFu + ((u >> 16) & 1u);
    return (unsigned short)(u >> 16);
}

// ---------------------------------------------------------------------------
// TDM: DMA one 128x32 bf16 tile (row stride ld elems) into LDS with padding
// that yields an 80B LDS row stride (16 dwords data + 4 dwords pad).
// Descriptor per CDNA5 ISA section 8. Issued by one wave; EXEC ignored.
// ---------------------------------------------------------------------------
__device__ __forceinline__ void tdm_load_tile_128x32(
    const unsigned short* gptr,   // tile start (bf16 elements)
    unsigned ld,                  // row stride in elements
    unsigned lds_byte_off,        // LDS byte offset of destination
    unsigned rows_total,          // tensor dim1 (rows) for OOB fields
    unsigned cols_total)          // tensor dim0 (cols) for OOB fields
{
    unsigned long long ga = (unsigned long long)(uintptr_t)gptr;

    u32x4 g0;
    g0[0] = 1u;                                            // count=1, user desc
    g0[1] = lds_byte_off;                                  // lds_addr
    g0[2] = (unsigned)(ga & 0xFFFFFFFFull);                // global_addr lo
    g0[3] = (unsigned)((ga >> 32) & 0x01FFFFFFull)         // global_addr hi
          | (2u << 30);                                    // type=2 ("image")

    i32x8 g1;
    g1[0] = (int)((1u << 16)      // data_size = 2 bytes
                | (1u << 20)      // pad_enable
                | (3u << 22)      // pad_interval: every 16 dwords
                | (3u << 25));    // pad_amount: 4 dwords
    g1[1] = (int)((cols_total & 0xFFFFu) << 16);                       // dim0 lo16
    g1[2] = (int)(((cols_total >> 16) & 0xFFFFu)
                | ((rows_total & 0xFFFFu) << 16));                     // dim0 hi | dim1 lo
    g1[3] = (int)(((rows_total >> 16) & 0xFFFFu) | (32u << 16));       // dim1 hi | tile_dim0=32
    g1[4] = (int)128u;                                                 // tile_dim1=128, tile_dim2=0
    g1[5] = (int)ld;                                                   // dim0_stride lo32
    g1[6] = 0;                                                         // dim0_stride hi16 | dim1_stride lo
    g1[7] = 0;

    i32x4 z4 = {0, 0, 0, 0};
    i32x8 z8 = {0, 0, 0, 0, 0, 0, 0, 0};
    __builtin_amdgcn_tensor_load_to_lds(g0, g1, z4, z4, z8, 0);
}

// ---------------------------------------------------------------------------
// bf16-WMMA GEMM:  C[M,N] = alpha * A[M,K] * W[N,K]^T  (+C if acc)
// A, W bf16 row-major. Tiles staged by the Tensor Data Mover (double
// buffered). 256 threads = 8 waves; 128x128 tile; wave does 32(M) x 64(N).
// Optional fp32 output C and/or bf16 output Cb.
// Requires M,N % 128 == 0, K % 32 == 0.
// ---------------------------------------------------------------------------
__global__ __launch_bounds__(256)
void gemm_bf16_wmma(const unsigned short* __restrict__ A, int lda,
                    const unsigned short* __restrict__ W, int ldw,
                    float* __restrict__ C, unsigned short* __restrict__ Cb, int ldc,
                    int M, int N, int K, float alpha, int accumulate)
{
    __shared__ __align__(16) unsigned short As[2][TILE_M * LDS_STRIDE];
    __shared__ __align__(16) unsigned short Bs[2][TILE_N * LDS_STRIDE];

    const int tid  = threadIdx.x;
    const int lane = tid & 31;
    const int wid  = tid >> 5;
    const int wm   = (wid & 3) * 32;
    const int wn   = (wid >> 2) * 64;

    const int bm = blockIdx.y * TILE_M;
    const int bn = blockIdx.x * TILE_N;

    const unsigned ldsA0 = (unsigned)(uintptr_t)(void*)&As[0][0];
    const unsigned ldsA1 = (unsigned)(uintptr_t)(void*)&As[1][0];
    const unsigned ldsB0 = (unsigned)(uintptr_t)(void*)&Bs[0][0];
    const unsigned ldsB1 = (unsigned)(uintptr_t)(void*)&Bs[1][0];

    v8f acc[2][4];
#pragma unroll
    for (int i = 0; i < 2; ++i)
#pragma unroll
        for (int j = 0; j < 4; ++j) acc[i][j] = (v8f){};

    // prologue: DMA first K-tile into buffer 0
    if (wid == 0) {
        tdm_load_tile_128x32(A + (size_t)bm * lda, (unsigned)lda, ldsA0,
                             (unsigned)M, (unsigned)K);
        tdm_load_tile_128x32(W + (size_t)bn * ldw, (unsigned)ldw, ldsB0,
                             (unsigned)N, (unsigned)K);
        __builtin_amdgcn_s_wait_tensorcnt(0);
    }
    __syncthreads();

    int buf = 0;
    for (int kk = 0; kk < K; kk += TILE_K) {
        const int havenext = (kk + TILE_K < K);
        if (wid == 0 && havenext) {   // DMA next K-tile into other buffer
            tdm_load_tile_128x32(A + (size_t)bm * lda + (kk + TILE_K),
                                 (unsigned)lda, buf ? ldsA0 : ldsA1,
                                 (unsigned)M, (unsigned)K);
            tdm_load_tile_128x32(W + (size_t)bn * ldw + (kk + TILE_K),
                                 (unsigned)ldw, buf ? ldsB0 : ldsB1,
                                 (unsigned)N, (unsigned)K);
        }

        // ---- fragments from LDS (ISA 7.12.2 layouts) ----
        v16bf afrag[2];
#pragma unroll
        for (int i = 0; i < 2; ++i) {
            int r  = wm + i * 16 + (lane & 15);
            int kb = (lane < 16) ? 0 : 8;
            const unsigned short* p = &As[buf][r * LDS_STRIDE + kb];
            uint4* d = reinterpret_cast<uint4*>(&afrag[i]);
            d[0] = *(const uint4*)(p);        // K = kb .. kb+7
            d[1] = *(const uint4*)(p + 16);   // K = kb+16 .. kb+23
        }
        v16bf bfrag[4];
#pragma unroll
        for (int j = 0; j < 4; ++j) {
            int n  = wn + j * 16 + (lane & 15);
            int kb = (lane < 16) ? 0 : 8;
            const unsigned short* p = &Bs[buf][n * LDS_STRIDE + kb];
            uint4* d = reinterpret_cast<uint4*>(&bfrag[j]);
            d[0] = *(const uint4*)(p);
            d[1] = *(const uint4*)(p + 16);
        }

        // ---- 8 WMMAs per wave per K-step ----
#pragma unroll
        for (int i = 0; i < 2; ++i)
#pragma unroll
            for (int j = 0; j < 4; ++j)
                acc[i][j] = __builtin_amdgcn_wmma_f32_16x16x32_bf16(
                    false, afrag[i], false, bfrag[j], (short)0, acc[i][j], false, false);

        if (wid == 0 && havenext) __builtin_amdgcn_s_wait_tensorcnt(0);
        __syncthreads();
        buf ^= 1;
    }

    // ---- epilogue: C layout lane 0-15 -> M=v, lanes 16-31 -> M=v+8 ----
#pragma unroll
    for (int i = 0; i < 2; ++i) {
#pragma unroll
        for (int j = 0; j < 4; ++j) {
            int n     = bn + wn + j * 16 + (lane & 15);
            int rbase = bm + wm + i * 16 + ((lane < 16) ? 0 : 8);
#pragma unroll
            for (int v = 0; v < 8; ++v) {
                size_t idx = (size_t)(rbase + v) * ldc + n;
                float  val = alpha * acc[i][j][v];
                if (C) {
                    if (accumulate) val += C[idx];
                    C[idx] = val;
                }
                if (Cb) Cb[idx] = f2bf(val);
            }
        }
    }
}

// ---------------------------------------------------------------------------
// Streaming fp32 -> bf16 conversion (one float4 per thread).
// ---------------------------------------------------------------------------
__global__ __launch_bounds__(256)
void convert_f32_bf16(const float* __restrict__ in, unsigned short* __restrict__ out)
{
    size_t i = ((size_t)blockIdx.x * 256 + threadIdx.x) * 4;
    float4 v = *(const float4*)(in + i);
    unsigned lo = (unsigned)f2bf(v.x) | ((unsigned)f2bf(v.y) << 16);
    unsigned hi = (unsigned)f2bf(v.z) | ((unsigned)f2bf(v.w) << 16);
    *(uint2*)(out + i) = make_uint2(lo, hi);
}

// ---------------------------------------------------------------------------
// Fused top-K(32) + softmax + weighted gather of mem_values (writes bf16 rp).
// One 256-thread block per token; sim row (4096 f32) kept in LDS.
// ---------------------------------------------------------------------------
#define N_SLOTS 4096
#define TOP_K   32
#define D_MODEL 2048

__global__ __launch_bounds__(256)
void topk_softmax_retrieve(const float* __restrict__ sim,        // [chunkT, N_SLOTS]
                           const float* __restrict__ mem_values, // [N_SLOTS, D]
                           float* __restrict__ w_out,            // [BT, K]
                           float* __restrict__ idx_out,          // [BT, K] (as float)
                           unsigned short* __restrict__ rpb,     // [BT, D] bf16
                           int token_base)
{
    const int t_local = blockIdx.x;
    const int t       = token_base + t_local;
    const int tid     = threadIdx.x;

    __shared__ float s[N_SLOTS];
    __shared__ float redv[256];
    __shared__ int   redi[256];
    __shared__ float topv[TOP_K];
    __shared__ int   topi[TOP_K];
    __shared__ float topw[TOP_K];

    const float* srow = sim + (size_t)t_local * N_SLOTS;
#pragma unroll
    for (int i = 0; i < 16; ++i) s[tid + 256 * i] = srow[tid + 256 * i];
    __syncthreads();

    for (int k = 0; k < TOP_K; ++k) {
        float bv = -INFINITY; int bi = 0;
#pragma unroll
        for (int i = 0; i < 16; ++i) {
            int   idx = tid + 256 * i;
            float v   = s[idx];
            if (v > bv) { bv = v; bi = idx; }
        }
        redv[tid] = bv; redi[tid] = bi;
        __syncthreads();
        for (int off = 128; off > 0; off >>= 1) {
            if (tid < off && redv[tid + off] > redv[tid]) {
                redv[tid] = redv[tid + off]; redi[tid] = redi[tid + off];
            }
            __syncthreads();
        }
        if (tid == 0) {
            topv[k] = redv[0]; topi[k] = redi[0];
            s[redi[0]] = -INFINITY;
        }
        __syncthreads();
    }

    // prefetch the gathered value rows as soon as indices are known
    if (tid < TOP_K) {
        const float* vrow = mem_values + (size_t)topi[tid] * D_MODEL;
        __builtin_prefetch(vrow, 0, 3);
        __builtin_prefetch(vrow + 1024, 0, 3);
    }

    if (tid == 0) {  // softmax over 32
        float m = topv[0];
        for (int k = 1; k < TOP_K; ++k) m = fmaxf(m, topv[k]);
        float sum = 0.f;
        for (int k = 0; k < TOP_K; ++k) { float e = __expf(topv[k] - m); topw[k] = e; sum += e; }
        float inv = 1.0f / sum;
        for (int k = 0; k < TOP_K; ++k) topw[k] *= inv;
    }
    __syncthreads();

    if (tid < TOP_K) {
        w_out[(size_t)t * TOP_K + tid]   = topw[tid];
        idx_out[(size_t)t * TOP_K + tid] = (float)topi[tid];
    }

    float accr[8];
#pragma unroll
    for (int i = 0; i < 8; ++i) accr[i] = 0.f;
    for (int k = 0; k < TOP_K; ++k) {
        const float* vrow = mem_values + (size_t)topi[k] * D_MODEL;
        float w = topw[k];
#pragma unroll
        for (int i = 0; i < 8; ++i) accr[i] += w * vrow[tid + 256 * i];
    }
#pragma unroll
    for (int i = 0; i < 8; ++i)
        rpb[(size_t)t * D_MODEL + tid + 256 * i] = f2bf(accr[i]);
}

// ---------------------------------------------------------------------------
// Gate epilogue: gate = sigmoid(wg2 . gelu(h + bg1) + bg2);  out = x + gate*rpf
// ---------------------------------------------------------------------------
__global__ __launch_bounds__(256)
void gate_output(const float* __restrict__ x, const float* __restrict__ rpf,
                 const float* __restrict__ h, const float* __restrict__ bg1,
                 const float* __restrict__ wg2, const float* __restrict__ bg2,
                 float* __restrict__ out)
{
    const int t   = blockIdx.x;
    const int tid = threadIdx.x;
    __shared__ float red[256];

    float part = 0.f;
#pragma unroll
    for (int i = 0; i < 4; ++i) {
        int   j  = tid + 256 * i;
        float hv = h[(size_t)t * 1024 + j] + bg1[j];
        float g  = 0.5f * hv * (1.0f + erff(hv * 0.70710678118654752f));
        part += g * wg2[j];
    }
    red[tid] = part;
    __syncthreads();
    for (int off = 128; off > 0; off >>= 1) {
        if (tid < off) red[tid] += red[tid + off];
        __syncthreads();
    }
    float gate = 1.0f / (1.0f + __expf(-(red[0] + bg2[0])));

#pragma unroll
    for (int i = 0; i < 8; ++i) {
        size_t o = (size_t)t * D_MODEL + tid + 256 * i;
        out[o] = x[o] + gate * rpf[o];
    }
}

// ---------------------------------------------------------------------------
// Launch
// ---------------------------------------------------------------------------
extern "C" void kernel_launch(void* const* d_in, const int* in_sizes, int n_in,
                              void* d_out, int out_size, void* d_ws, size_t ws_size,
                              hipStream_t stream) {
    const float* x          = (const float*)d_in[0];
    const float* mem_keys   = (const float*)d_in[1];
    const float* mem_values = (const float*)d_in[2];
    const float* wq         = (const float*)d_in[3];
    const float* wo         = (const float*)d_in[4];
    const float* wg1        = (const float*)d_in[5];
    const float* bg1        = (const float*)d_in[6];
    const float* wg2        = (const float*)d_in[7];
    const float* bg2        = (const float*)d_in[8];
    float* out = (float*)d_out;

    const int BT = 8192, D = 2048, NS = 4096, DH = 1024, TK = 32;
    const size_t MB = 1024 * 1024;

    char* ws = (char*)d_ws;
    unsigned short* xb   = (unsigned short*)(ws);              // 32MB [BT,D]
    unsigned short* qb   = (unsigned short*)(ws + 32  * MB);   // 32MB [BT,D]
    unsigned short* rpb  = (unsigned short*)(ws + 64  * MB);   // 32MB [BT,D]
    unsigned short* rpfb = (unsigned short*)(ws + 96  * MB);   // 32MB [BT,D]
    float*          rpf  = (float*)         (ws + 128 * MB);   // 64MB [BT,D]
    float*          h    = (float*)         (ws + 192 * MB);   // 32MB [BT,DH]
    float*          sim  = (float*)         (ws + 224 * MB);   // 32MB [2048,NS]
    unsigned short* wqb  = (unsigned short*)(ws + 256 * MB);   //  8MB [D,D]
    unsigned short* mkb  = (unsigned short*)(ws + 264 * MB);   // 16MB [NS,D]
    unsigned short* wob  = (unsigned short*)(ws + 280 * MB);   //  8MB [D,D]
    unsigned short* wg1b = (unsigned short*)(ws + 288 * MB);   //  8MB [DH,2D]

    float* w_out = out + (size_t)BT * D;       // top_w  [BT,32]
    float* i_out = w_out + (size_t)BT * TK;    // top_idx[BT,32] (as float)

    dim3 blk(256);

    // 0) convert fp32 operands to bf16 (streaming; ~216MB read+written, negligible)
    convert_f32_bf16<<<dim3((unsigned)((size_t)BT * D / 1024)), blk, 0, stream>>>(x, xb);
    convert_f32_bf16<<<dim3((unsigned)((size_t)D  * D / 1024)), blk, 0, stream>>>(wq, wqb);
    convert_f32_bf16<<<dim3((unsigned)((size_t)NS * D / 1024)), blk, 0, stream>>>(mem_keys, mkb);
    convert_f32_bf16<<<dim3((unsigned)((size_t)D  * D / 1024)), blk, 0, stream>>>(wo, wob);
    convert_f32_bf16<<<dim3((unsigned)((size_t)DH * 2 * D / 1024)), blk, 0, stream>>>(wg1, wg1b);

    // 1) qb = bf16(x @ wq^T)
    gemm_bf16_wmma<<<dim3(D / 128, BT / 128), blk, 0, stream>>>(
        xb, D, wqb, D, nullptr, qb, D, BT, D, D, 1.0f, 0);

    // 2) sim = q @ mem_keys^T / sqrt(D), chunked; fused topk+softmax+gather
    const float scale = 0.022097086912079612f;  // 1/sqrt(2048)
    const int CH = 2048;
    for (int c = 0; c < BT / CH; ++c) {
        gemm_bf16_wmma<<<dim3(NS / 128, CH / 128), blk, 0, stream>>>(
            qb + (size_t)c * CH * D, D, mkb, D, sim, nullptr, NS, CH, NS, D, scale, 0);
        topk_softmax_retrieve<<<dim3(CH), blk, 0, stream>>>(
            sim, mem_values, w_out, i_out, rpb, c * CH);
    }

    // 3) rpf = rp @ wo^T  (fp32 + bf16 outputs)
    gemm_bf16_wmma<<<dim3(D / 128, BT / 128), blk, 0, stream>>>(
        rpb, D, wob, D, rpf, rpfb, D, BT, D, D, 1.0f, 0);

    // 4) h = x @ wg1[:, :D]^T + rpf @ wg1[:, D:]^T   (split-concat, accumulating)
    gemm_bf16_wmma<<<dim3(DH / 128, BT / 128), blk, 0, stream>>>(
        xb, D, wg1b, 2 * D, h, nullptr, DH, BT, DH, D, 1.0f, 0);
    gemm_bf16_wmma<<<dim3(DH / 128, BT / 128), blk, 0, stream>>>(
        rpfb, D, wg1b + D, 2 * D, h, nullptr, DH, BT, DH, D, 1.0f, 1);

    // 5) gate + residual output
    gate_output<<<dim3(BT), blk, 0, stream>>>(x, rpf, h, bg1, wg2, bg2, out);
}
